// DeformConv2D_2267742732651
// MI455X (gfx1250) — matrix operand order repaired and checked by reading, not compiled, not run on hardware
//
#include <hip/hip_runtime.h>

typedef float v8f  __attribute__((ext_vector_type(8)));
typedef __bf16 v16bf __attribute__((ext_vector_type(16)));
typedef unsigned int u32x2 __attribute__((ext_vector_type(2)));
typedef unsigned int u32x4 __attribute__((ext_vector_type(4)));

#define HIN 64
#define WIN 64
#define CIN 256
#define COUT 256
#define KK 9
#define KDIM (KK * CIN)        // 2304
#define KPAD 2312              // padded LDS row stride (halves) to avoid bank conflicts
#define HP 66                  // padded spatial extent
#define OFFC 18
#define MROWS 32               // output positions per workgroup (2 WMMA M-subtiles)

__device__ __forceinline__ unsigned short f2bf(float f) {
    union { __bf16 b; unsigned short s; } u;
    u.b = (__bf16)f;           // round-to-nearest-even
    return u.s;
}

// ---------------- Kernel 0: weights HWIO f32 -> Wt[co][k] bf16 ----------------
__global__ __launch_bounds__(256) void wt_prep(const float* __restrict__ W,
                                               unsigned short* __restrict__ Wt) {
    int idx = blockIdx.x * 256 + threadIdx.x;   // grid = KDIM, coalesced reads
    int k  = idx >> 8;                          // 0..2303
    int co = idx & 255;
    Wt[(size_t)co * KDIM + k] = f2bf(W[idx]);
}

// ---------------- Kernel 1: offset conv (3x3, C=256 -> 18) ----------------
__global__ __launch_bounds__(128) void offset_conv(const float* __restrict__ inp,
                                                   const float* __restrict__ oW,
                                                   const float* __restrict__ oB,
                                                   float* __restrict__ offbuf) {
    const int wid  = threadIdx.x >> 5;
    const int lane = threadIdx.x & 31;
    const int m = blockIdx.x * 4 + wid;         // grid = 4096 -> 16384 positions
    const int b = m >> 12;
    const int y = (m >> 6) & 63;
    const int x = m & 63;

    float acc[OFFC];
#pragma unroll
    for (int j = 0; j < OFFC; ++j) acc[j] = 0.f;

    const int cbase = lane * 8;                 // 8 channels per lane
#pragma unroll
    for (int ky = 0; ky < 3; ++ky) {
        int iy = y + ky - 1;
        if (iy < 0 || iy > HIN - 1) continue;   // zero padding
#pragma unroll
        for (int kx = 0; kx < 3; ++kx) {
            int ix = x + kx - 1;
            if (ix < 0 || ix > WIN - 1) continue;
            const float* ip = inp + ((size_t)((b * HIN + iy) * WIN + ix)) * CIN + cbase;
            float v[8];
#pragma unroll
            for (int ci = 0; ci < 8; ++ci) v[ci] = ip[ci];
            const float* wbase = oW + ((size_t)((ky * 3 + kx) * CIN + cbase)) * OFFC;
#pragma unroll
            for (int ci = 0; ci < 8; ++ci) {
                const float* wp = wbase + ci * OFFC;
#pragma unroll
                for (int j = 0; j < OFFC; ++j) acc[j] = fmaf(v[ci], wp[j], acc[j]);
            }
        }
    }
    // wave32 cross-lane reduction
#pragma unroll
    for (int j = 0; j < OFFC; ++j) {
        float a = acc[j];
        for (int off = 16; off > 0; off >>= 1) a += __shfl_down(a, off, 32);
        if (lane == 0) offbuf[(size_t)m * OFFC + j] = a + oB[j];
    }
}

// ---------------- Kernel 2: bilinear gather + WMMA GEMM ----------------
__global__ __launch_bounds__(128) void deform_main(const float* __restrict__ inp,
                                                   const unsigned short* __restrict__ Wt,
                                                   const float* __restrict__ bias,
                                                   const float* __restrict__ offbuf,
                                                   float* __restrict__ out) {
    // 32 rows x 2304 (padded to 2312) bf16 = 147,968 B LDS
    __shared__ __align__(16) unsigned short As[MROWS * KPAD];

    const int tid = threadIdx.x;
    const int wg  = blockIdx.x;                 // M-tile index (32 positions)

    // ---- Stage 1: sample A tile into LDS (bf16) ----
    {
        const int cg = tid & 63;                // channel group: 4 channels
        const int ph = tid >> 6;                // 0..1
        const int c  = cg * 4;
        for (int pp = ph; pp < MROWS * KK; pp += 2) {
            const int p  = pp / KK;
            const int kk = pp % KK;
            const int m  = wg * MROWS + p;
            const int b  = m >> 12;
            const int y  = (m >> 6) & 63;
            const int x  = m & 63;

            const size_t ob = (size_t)m * OFFC + kk * 2;
            float yo = offbuf[ob], xo = offbuf[ob + 1];
            float yp = (float)(y + kk / 3) + yo;          // padded coords
            float xp = (float)(x + kk % 3) + xo;
            yp = fminf(fmaxf(yp, 0.f), (float)(HP - 1));
            xp = fminf(fmaxf(xp, 0.f), (float)(HP - 1));
            int y0 = (int)floorf(yp), x0 = (int)floorf(xp);
            int y0c = min(max(y0, 0), HP - 1), y1c = min(max(y0 + 1, 0), HP - 1);
            int x0c = min(max(x0, 0), HP - 1), x1c = min(max(x0 + 1, 0), HP - 1);
            float ly = yp - (float)y0c, hy = (float)y1c - yp;
            float lx = xp - (float)x0c, hx = (float)x1c - xp;
            float w0 = hy * hx, w1 = hy * lx, w2 = ly * hx, w3 = ly * lx;

            float4 v00 = {0,0,0,0}, v01 = {0,0,0,0}, v10 = {0,0,0,0}, v11 = {0,0,0,0};
            bool vy0 = (y0c >= 1) && (y0c <= HIN);
            bool vy1 = (y1c >= 1) && (y1c <= HIN);
            bool vx0 = (x0c >= 1) && (x0c <= WIN);
            bool vx1 = (x1c >= 1) && (x1c <= WIN);
            const size_t bb = (size_t)b * HIN * WIN * CIN;
            if (vy0 && vx0) v00 = *(const float4*)(inp + bb + ((size_t)((y0c - 1) * WIN + (x0c - 1))) * CIN + c);
            if (vy0 && vx1) v01 = *(const float4*)(inp + bb + ((size_t)((y0c - 1) * WIN + (x1c - 1))) * CIN + c);
            if (vy1 && vx0) v10 = *(const float4*)(inp + bb + ((size_t)((y1c - 1) * WIN + (x0c - 1))) * CIN + c);
            if (vy1 && vx1) v11 = *(const float4*)(inp + bb + ((size_t)((y1c - 1) * WIN + (x1c - 1))) * CIN + c);

            float r0 = w0 * v00.x + w1 * v01.x + w2 * v10.x + w3 * v11.x;
            float r1 = w0 * v00.y + w1 * v01.y + w2 * v10.y + w3 * v11.y;
            float r2 = w0 * v00.z + w1 * v01.z + w2 * v10.z + w3 * v11.z;
            float r3 = w0 * v00.w + w1 * v01.w + w2 * v10.w + w3 * v11.w;

            u32x2 pk;
            pk.x = (unsigned int)f2bf(r0) | ((unsigned int)f2bf(r1) << 16);
            pk.y = (unsigned int)f2bf(r2) | ((unsigned int)f2bf(r3) << 16);
            *reinterpret_cast<u32x2*>(&As[p * KPAD + kk * CIN + c]) = pk;
        }
    }
    __syncthreads();

    // ---- Stage 2: WMMA over K = 2304 (72 steps of 32) ----
    // Per wave: 2 M-subtiles x 4 N-tiles = 8 accumulators; each B fragment
    // feeds 2 WMMAs -> 1.5 b128 loads per WMMA.
    const int lane = tid & 31;
    const int wid  = tid >> 5;
    const int mrow = lane & 15;
    const int half = lane >> 4;
    const int co_base = wid * 64;

    v8f acc[2][4];
#pragma unroll
    for (int s = 0; s < 2; ++s)
#pragma unroll
        for (int nt = 0; nt < 4; ++nt)
            acc[s][nt] = (v8f){0.f,0.f,0.f,0.f,0.f,0.f,0.f,0.f};

    for (int kb = 0; kb < KDIM / 32; ++kb) {
        // A fragments: lane L -> M = L%16 (+16 for subtile 1); elements 0..7 at
        // K = kb*32 + (L/16)*8, elements 8..15 at +16 (16-bit A 16x32 layout).
        union { u32x4 u[2]; v16bf v; } af[2];
#pragma unroll
        for (int s = 0; s < 2; ++s) {
            const int aoff = (s * 16 + mrow) * KPAD + kb * 32 + half * 8;
            af[s].u[0] = *reinterpret_cast<const u32x4*>(&As[aoff]);
            af[s].u[1] = *reinterpret_cast<const u32x4*>(&As[aoff + 16]);
        }

        const int k0 = kb * 32 + half * 8;      // B: lane L -> N = L%16, K = (L/16)*8 + e (e<8), +16 for e>=8
#pragma unroll
        for (int nt = 0; nt < 4; ++nt) {
            const unsigned short* wp = Wt + (size_t)(co_base + nt * 16 + mrow) * KDIM + k0;
            union { u32x4 u[2]; v16bf v; } bf_;
            bf_.u[0] = *reinterpret_cast<const u32x4*>(wp);
            bf_.u[1] = *reinterpret_cast<const u32x4*>(wp + 16);
            acc[0][nt] = __builtin_amdgcn_wmma_f32_16x16x32_bf16(
                false, af[0].v, false, bf_.v, (short)0, acc[0][nt], false, false);
            acc[1][nt] = __builtin_amdgcn_wmma_f32_16x16x32_bf16(
                false, af[1].v, false, bf_.v, (short)0, acc[1][nt], false, false);
        }
    }

    // ---- Epilogue: C/D layout VGPR r -> M = r + 8*half, N = lane%16 ----
#pragma unroll
    for (int nt = 0; nt < 4; ++nt) {
        const int co = co_base + nt * 16 + mrow;
        const float bsum = bias[co];
#pragma unroll
        for (int s = 0; s < 2; ++s) {
#pragma unroll
            for (int r = 0; r < 8; ++r) {
                const int m = wg * MROWS + s * 16 + r + half * 8;
                out[(size_t)m * COUT + co] = acc[s][nt][r] + bsum;
            }
        }
    }
}

extern "C" void kernel_launch(void* const* d_in, const int* in_sizes, int n_in,
                              void* d_out, int out_size, void* d_ws, size_t ws_size,
                              hipStream_t stream) {
    const float* inp  = (const float*)d_in[0];
    const float* W    = (const float*)d_in[1];
    const float* bias = (const float*)d_in[2];
    const float* oW   = (const float*)d_in[3];
    const float* oB   = (const float*)d_in[4];
    float* out = (float*)d_out;

    unsigned short* Wt = (unsigned short*)d_ws;                       // 256*2304*2 = 1,179,648 B
    float* offbuf = (float*)((char*)d_ws + (size_t)COUT * KDIM * 2);  // 4*64*64*18*4 = 1,179,648 B

    wt_prep<<<KDIM, 256, 0, stream>>>(W, Wt);
    offset_conv<<<4096, 128, 0, stream>>>(inp, oW, oB, offbuf);
    deform_main<<<16384 / MROWS, 128, 0, stream>>>(inp, Wt, bias, offbuf, out);
}
